// MultiHeadSelfAttention_11493332484903
// MI455X (gfx1250) — matrix operand order repaired and checked by reading, not compiled, not run on hardware
//
#include <hip/hip_runtime.h>

typedef __attribute__((ext_vector_type(16))) __bf16 v16bf;
typedef __attribute__((ext_vector_type(8)))  __bf16 v8bf;
typedef __attribute__((ext_vector_type(8)))  float  v8f;

#define B_  2
#define S_  2048
#define D_  1024
#define H_  16
#define DH_ 64

// ---------------- conversion kernels ----------------

__global__ __launch_bounds__(256) void cvt_f32_to_bf16(const float* __restrict__ src,
                                                       __bf16* __restrict__ dst, int n) {
  int i = blockIdx.x * blockDim.x + threadIdx.x;
  if (i < n) dst[i] = (__bf16)src[i];
}

// W [D,D] fp32 -> Wt [n][k] bf16 (transposed), so WMMA B-fragments are contiguous.
__global__ __launch_bounds__(256) void cvt_transpose_w(const float* __restrict__ W,
                                                       __bf16* __restrict__ Wt) {
  int i = blockIdx.x * blockDim.x + threadIdx.x; // over D_*D_
  int k = i / D_, n = i % D_;
  Wt[(size_t)n * D_ + k] = (__bf16)W[(size_t)k * D_ + n];
}

// Async global->LDS copy of one 16-byte chunk per lane (one instruction per wave).
__device__ __forceinline__ void async_copy_b128(unsigned lds_byte_addr, const void* gaddr) {
  asm volatile("global_load_async_to_lds_b128 %0, %1, off"
               :: "v"(lds_byte_addr), "v"(gaddr) : "memory");
}
__device__ __forceinline__ void wait_asynccnt0() {
  asm volatile("s_wait_asynccnt 0" ::: "memory");
}

// ---------------- WMMA GEMM: out = A[M,K] @ Bt[N,K]^T + bias ----------------
// B tile (64 cols x 32 k) is shared by all 8 waves -> staged in LDS via async
// load-to-LDS DMA, double buffered. A fragments are per-wave-unique direct loads.
// mode 0: store bf16 row-major [M,N]
// mode 1: store bf16 as V-transposed [B,H,DH,S]   (m=b*S+s, n=h*DH+dh)
// mode 2: store fp32 row-major [M,N]
__global__ __launch_bounds__(256) void gemm_bf16_wmma(
    const __bf16* __restrict__ A,
    const __bf16* __restrict__ Bt,
    const float*  __restrict__ bias,
    void* __restrict__ out, int M, int N, int K, int mode)
{
  __shared__ __align__(32) __bf16 ldsB[2][64 * 32];   // 2 x 4 KB double buffer

  int tid  = threadIdx.x;
  int lane = tid & 31;
  int wave = tid >> 5;
  int half = lane >> 4;   // 0 or 1
  int l16  = lane & 15;

  int m0 = blockIdx.y * 128 + wave * 16;
  int n0 = blockIdx.x * 64;

  // per-thread async-copy assignment: 256 threads x 16B = 4 KB tile
  int prow = tid >> 2;          // 0..63  (column index within N-strip)
  int pcol = (tid & 3) * 8;     // element offset 0,8,16,24 within 32-wide K chunk
  const __bf16* gB = Bt + (size_t)(n0 + prow) * K + pcol;
  unsigned ldsBase = (unsigned)(unsigned long long)(uintptr_t)&ldsB[0][0];
  unsigned ldsSlot = ldsBase + (unsigned)(prow * 64 + pcol * 2); // bytes
  const unsigned BUFSZ = 64 * 32 * 2;                            // 4096 bytes

  async_copy_b128(ldsSlot, gB);         // prefetch kc = 0 into buffer 0

  v8f acc[4] = {};
  const __bf16* Arow = A + (size_t)(m0 + l16) * K;

  int buf = 0;
  for (int kc = 0; kc < K; kc += 32) {
    // A fragment first (long-latency global load overlaps barrier/LDS reads)
    v16bf a;
    ((v8bf*)&a)[0] = *(const v8bf*)(Arow + kc + half * 16);
    ((v8bf*)&a)[1] = *(const v8bf*)(Arow + kc + half * 16 + 8);

    wait_asynccnt0();
    __syncthreads();                    // buffer `buf` ready for all waves
    if (kc + 32 < K)
      async_copy_b128(ldsSlot + (unsigned)((buf ^ 1) * BUFSZ), gB + kc + 32);

    // gather all four B fragments, then issue the four WMMAs back-to-back
    v16bf bfrag[4];
#pragma unroll
    for (int sub = 0; sub < 4; ++sub)
      bfrag[sub] = *(const v16bf*)&ldsB[buf][(sub * 16 + l16) * 32 + half * 16];
#pragma unroll
    for (int sub = 0; sub < 4; ++sub)
      acc[sub] = __builtin_amdgcn_wmma_f32_16x16x32_bf16(
          false, a, false, bfrag[sub], (short)0, acc[sub], false, false);

    __syncthreads();                    // all waves done reading `buf`
    buf ^= 1;
  }

#pragma unroll
  for (int sub = 0; sub < 4; ++sub) {
    int n = n0 + sub * 16 + l16;
    float bb = bias[n];
#pragma unroll
    for (int r = 0; r < 8; ++r) {
      int m = m0 + r + half * 8;
      float v = acc[sub][r] + bb;
      if (mode == 0) {
        ((__bf16*)out)[(size_t)m * N + n] = (__bf16)v;
      } else if (mode == 1) {
        int b_ = m >> 11, s = m & (S_ - 1);   // S_ = 2048
        int h_ = n >> 6,  dh = n & (DH_ - 1);
        ((__bf16*)out)[(((size_t)(b_ * H_ + h_)) * DH_ + dh) * S_ + s] = (__bf16)v;
      } else {
        ((float*)out)[(size_t)m * N + n] = v;
      }
    }
  }
}

// ---------------- flash attention: one wave per (b, h, 16-query tile) ----------------
__global__ __launch_bounds__(32) void attention_wmma(
    const __bf16* __restrict__ Q,   // [B,S,D]  (per-head slice contiguous over DH)
    const __bf16* __restrict__ Kb,  // [B,S,D]
    const __bf16* __restrict__ Vt,  // [B,H,DH,S]
    __bf16* __restrict__ O)         // [B,S,D]
{
  __shared__ __align__(32) __bf16 ldsP[16 * 32];

  int lane = threadIdx.x & 31;
  int half = lane >> 4;
  int l16  = lane & 15;
  int q0 = blockIdx.x * 16;
  int h  = blockIdx.y;
  int b  = blockIdx.z;

  // Q fragments for the two 32-wide head-dim chunks, A layout, kept in registers.
  const __bf16* Qbase = Q + ((size_t)(b * S_ + q0 + l16)) * D_ + h * DH_;
  v16bf aq[2];
#pragma unroll
  for (int kc = 0; kc < 2; ++kc) {
    ((v8bf*)&aq[kc])[0] = *(const v8bf*)(Qbase + kc * 32 + half * 16);
    ((v8bf*)&aq[kc])[1] = *(const v8bf*)(Qbase + kc * 32 + half * 16 + 8);
  }

  float mrow[8], lrow[8];
#pragma unroll
  for (int r = 0; r < 8; ++r) { mrow[r] = -1e30f; lrow[r] = 0.f; }
  v8f o[4] = {};

  const float scale = 0.125f; // 1/sqrt(DH)
  const __bf16* Kh = Kb + (size_t)b * S_ * D_ + h * DH_;
  const __bf16* Vh = Vt + ((size_t)(b * H_ + h)) * DH_ * S_;

  for (int c = 0; c < S_; c += 32) {
    // ---- scores: two 16x16 tiles (keys c..c+15 and c+16..c+31) ----
    // load all four K fragments, then issue the four WMMAs
    const __bf16* Kc0 = Kh + (size_t)(c + l16) * D_ + half * 16;
    const __bf16* Kc1 = Kh + (size_t)(c + 16 + l16) * D_ + half * 16;
    v16bf bk00 = *(const v16bf*)(Kc0);
    v16bf bk01 = *(const v16bf*)(Kc0 + 32);
    v16bf bk10 = *(const v16bf*)(Kc1);
    v16bf bk11 = *(const v16bf*)(Kc1 + 32);

    v8f s0 = {}, s1 = {};
    s0 = __builtin_amdgcn_wmma_f32_16x16x32_bf16(false, aq[0], false, bk00, (short)0, s0, false, false);
    s1 = __builtin_amdgcn_wmma_f32_16x16x32_bf16(false, aq[0], false, bk10, (short)0, s1, false, false);
    s0 = __builtin_amdgcn_wmma_f32_16x16x32_bf16(false, aq[1], false, bk01, (short)0, s0, false, false);
    s1 = __builtin_amdgcn_wmma_f32_16x16x32_bf16(false, aq[1], false, bk11, (short)0, s1, false, false);

    // ---- online softmax (row = r + 8*half, identical mapping for scores and O) ----
    float p0v[8], p1v[8];
#pragma unroll
    for (int r = 0; r < 8; ++r) {
      float a0 = s0[r] * scale, a1 = s1[r] * scale;
      float t = fmaxf(a0, a1);
      t = fmaxf(t, __shfl_xor(t, 1, 32));
      t = fmaxf(t, __shfl_xor(t, 2, 32));
      t = fmaxf(t, __shfl_xor(t, 4, 32));
      t = fmaxf(t, __shfl_xor(t, 8, 32));
      float mn  = fmaxf(mrow[r], t);
      float fac = __expf(mrow[r] - mn);
      float p0  = __expf(a0 - mn);
      float p1  = __expf(a1 - mn);
      float rs  = p0 + p1;
      rs += __shfl_xor(rs, 1, 32);
      rs += __shfl_xor(rs, 2, 32);
      rs += __shfl_xor(rs, 4, 32);
      rs += __shfl_xor(rs, 8, 32);
      lrow[r] = lrow[r] * fac + rs;
      mrow[r] = mn;
#pragma unroll
      for (int sub = 0; sub < 4; ++sub) o[sub][r] = o[sub][r] * fac;
      p0v[r] = p0; p1v[r] = p1;
    }

    // ---- transpose P (C layout -> A layout) through LDS ----
    __syncthreads();
#pragma unroll
    for (int r = 0; r < 8; ++r) {
      int row = r + half * 8;
      ldsP[row * 32 + l16]      = (__bf16)p0v[r];
      ldsP[row * 32 + 16 + l16] = (__bf16)p1v[r];
    }
    __syncthreads();

    v16bf ap;
    ((v8bf*)&ap)[0] = *(const v8bf*)(&ldsP[l16 * 32 + half * 16]);
    ((v8bf*)&ap)[1] = *(const v8bf*)(&ldsP[l16 * 32 + half * 16 + 8]);

    // ---- O += P x V (K = 32 keys, N = 4 x 16 head dims) ----
    v16bf bv[4];
#pragma unroll
    for (int sub = 0; sub < 4; ++sub)
      bv[sub] = *(const v16bf*)(Vh + (size_t)(sub * 16 + l16) * S_ + c + half * 16);
#pragma unroll
    for (int sub = 0; sub < 4; ++sub)
      o[sub] = __builtin_amdgcn_wmma_f32_16x16x32_bf16(
          false, ap, false, bv[sub], (short)0, o[sub], false, false);
  }

  // ---- normalize and store ----
#pragma unroll
  for (int r = 0; r < 8; ++r) {
    float inv = 1.0f / lrow[r];
    int row = r + half * 8;
    __bf16* Orow = O + ((size_t)(b * S_ + q0 + row)) * D_ + h * DH_;
#pragma unroll
    for (int sub = 0; sub < 4; ++sub)
      Orow[sub * 16 + l16] = (__bf16)(o[sub][r] * inv);
  }
}

// ---------------- host launcher ----------------

extern "C" void kernel_launch(void* const* d_in, const int* in_sizes, int n_in,
                              void* d_out, int out_size, void* d_ws, size_t ws_size,
                              hipStream_t stream) {
  (void)in_sizes; (void)n_in; (void)out_size; (void)ws_size;

  const float* x  = (const float*)d_in[0];
  const float* Wq = (const float*)d_in[1];
  const float* bq = (const float*)d_in[2];
  const float* Wk = (const float*)d_in[3];
  const float* bk = (const float*)d_in[4];
  const float* Wv = (const float*)d_in[5];
  const float* bv = (const float*)d_in[6];
  const float* Wo = (const float*)d_in[7];
  const float* bo = (const float*)d_in[8];
  float* out = (float*)d_out;

  char* ws = (char*)d_ws;
  const size_t MB = (size_t)1 << 20;
  __bf16* xb  = (__bf16*)(ws + 0 * MB);   // 8 MB : x in bf16 [B*S, D]
  __bf16* WqT = (__bf16*)(ws + 8 * MB);   // 2 MB each, transposed bf16 weights
  __bf16* WkT = (__bf16*)(ws + 10 * MB);
  __bf16* WvT = (__bf16*)(ws + 12 * MB);
  __bf16* WoT = (__bf16*)(ws + 14 * MB);
  __bf16* Qb  = (__bf16*)(ws + 16 * MB);  // 8 MB [B,S,D]
  __bf16* Kbf = (__bf16*)(ws + 24 * MB);  // 8 MB [B,S,D]
  __bf16* Vtb = (__bf16*)(ws + 32 * MB);  // 8 MB [B,H,DH,S]
  __bf16* Ab  = (__bf16*)(ws + 40 * MB);  // 8 MB attention out [B,S,D]

  const int M  = B_ * S_;       // 4096
  const int nx = M * D_;        // 4M elements
  const int nw = D_ * D_;       // 1M elements

  cvt_f32_to_bf16<<<(nx + 255) / 256, 256, 0, stream>>>(x, xb, nx);
  cvt_transpose_w<<<(nw + 255) / 256, 256, 0, stream>>>(Wq, WqT);
  cvt_transpose_w<<<(nw + 255) / 256, 256, 0, stream>>>(Wk, WkT);
  cvt_transpose_w<<<(nw + 255) / 256, 256, 0, stream>>>(Wv, WvT);
  cvt_transpose_w<<<(nw + 255) / 256, 256, 0, stream>>>(Wo, WoT);

  dim3 ggrid(D_ / 64, M / 128); // 16 x 32 blocks of 256 threads
  gemm_bf16_wmma<<<ggrid, 256, 0, stream>>>(xb, WqT, bq, Qb,  M, D_, D_, 0);
  gemm_bf16_wmma<<<ggrid, 256, 0, stream>>>(xb, WkT, bk, Kbf, M, D_, D_, 0);
  gemm_bf16_wmma<<<ggrid, 256, 0, stream>>>(xb, WvT, bv, Vtb, M, D_, D_, 1);

  dim3 agrid(S_ / 16, H_, B_);  // 128 x 16 x 2 single-wave workgroups
  attention_wmma<<<agrid, 32, 0, stream>>>(Qb, Kbf, Vtb, Ab);

  gemm_bf16_wmma<<<ggrid, 256, 0, stream>>>(Ab, WoT, bo, out, M, D_, D_, 2);
}
